// AttentionLayer_45973329936963
// MI455X (gfx1250) — matrix operand order repaired and checked by reading, not compiled, not run on hardware
//
#include <hip/hip_runtime.h>

#define NB 8
#define LQ 2048
#define HD 1024
#define ED 64
#define NROWS (NB * LQ)

typedef __attribute__((ext_vector_type(16))) __bf16          v16bf;
typedef __attribute__((ext_vector_type(8)))  float           v8f;
typedef __attribute__((ext_vector_type(16))) unsigned short  u16x16;
typedef __attribute__((ext_vector_type(8)))  unsigned short  u16x8;
typedef __attribute__((ext_vector_type(4)))  unsigned short  u16x4;
typedef __attribute__((ext_vector_type(4)))  float           f32x4;

// Native fp32 -> bf16 (RNE) via the compiler's hardware convert path.
static __device__ __forceinline__ unsigned short f2bf(float f) {
  __bf16 h = (__bf16)f;
  return __builtin_bit_cast(unsigned short, h);
}

static __device__ __forceinline__ v16bf pack_bf16(const float* f) {
  v16bf r;
#pragma unroll
  for (int i = 0; i < 16; ++i) r[i] = (__bf16)f[i];
  return r;
}

static __device__ __forceinline__ v16bf combine8(u16x8 lo, u16x8 hi) {
  u16x16 u;
#pragma unroll
  for (int i = 0; i < 8; ++i) { u[i] = lo[i]; u[i + 8] = hi[i]; }
  return __builtin_bit_cast(v16bf, u);
}

// ---------------------------------------------------------------------------
// Kernel 0: one-shot Wq/Wk/Wv fp32 -> bf16 conversion (3 x 64x1024, 768 KB).
// ---------------------------------------------------------------------------
__global__ __launch_bounds__(256)
void cvt_w_kernel(const float* __restrict__ Wq, const float* __restrict__ Wk,
                  const float* __restrict__ Wv, unsigned short* __restrict__ Wb)
{
  const float* src = (blockIdx.y == 0) ? Wq : (blockIdx.y == 1) ? Wk : Wv;
  unsigned short* dst = Wb + (size_t)blockIdx.y * ED * HD;
  const int i = (blockIdx.x * 256 + threadIdx.x) * 4;
  f32x4 x = *(const f32x4*)(src + i);
  u16x4 o;
#pragma unroll
  for (int j = 0; j < 4; ++j) o[j] = f2bf(x[j]);
  *(u16x4*)(dst + i) = o;
}

// ---------------------------------------------------------------------------
// Kernel 1: fused q/k/v projection  proj[r,e] = sum_h x[r,h]*W[e,h] + b[e]
// One wave computes one 16x64 output tile via v_wmma_f32_16x16x32_bf16.
// All four B tiles are loaded before the WMMA burst so the loads clause and
// only one load-latency is exposed per K-step.
// ---------------------------------------------------------------------------
__global__ __launch_bounds__(128)
void proj_kernel(const float* __restrict__ q, const float* __restrict__ k,
                 const float* __restrict__ v,
                 const unsigned short* __restrict__ Wb,
                 const float* __restrict__ bq, const float* __restrict__ bk,
                 const float* __restrict__ bv,
                 unsigned short* __restrict__ qp, unsigned short* __restrict__ kp,
                 unsigned short* __restrict__ vp)
{
  const float* X; const float* Bb; unsigned short* Out;
  if (blockIdx.y == 0)      { X = q; Bb = bq; Out = qp; }
  else if (blockIdx.y == 1) { X = k; Bb = bk; Out = kp; }
  else                      { X = v; Bb = bv; Out = vp; }
  const unsigned short* W = Wb + (size_t)blockIdx.y * ED * HD;

  const int wave = threadIdx.x >> 5;
  const int lane = threadIdx.x & 31;
  const int m    = lane & 15;
  const int half = lane >> 4;
  const int row0 = (blockIdx.x * 4 + wave) * 16;

  const float* xrow = X + (size_t)(row0 + m) * HD;
  const unsigned short* wbase = W + (size_t)m * HD + half * 16;

  v8f z = {0.f, 0.f, 0.f, 0.f, 0.f, 0.f, 0.f, 0.f};
  v8f acc[4] = {z, z, z, z};

  for (int k0 = 0; k0 < HD; k0 += 32) {
    // ---- load phase: A (fp32, converted) + all 4 B tiles (bf16) ----
    float af[16];
    const float* p1 = xrow + k0 + half * 16;
    const float* p2 = xrow + k0 + half * 16 + 8;
    *(f32x4*)(af + 0)  = *(const f32x4*)(p1);
    *(f32x4*)(af + 4)  = *(const f32x4*)(p1 + 4);
    *(f32x4*)(af + 8)  = *(const f32x4*)(p2);
    *(f32x4*)(af + 12) = *(const f32x4*)(p2 + 4);

    v16bf b[4];
#pragma unroll
    for (int t = 0; t < 4; ++t) {
      // B operand: B[k][n] = W[n][k]; lane = n, K contiguous per half (bf16)
      const unsigned short* wrow = wbase + (size_t)(t * 16) * HD + k0;
      b[t] = combine8(*(const u16x8*)wrow, *(const u16x8*)(wrow + 8));
    }
    v16bf a = pack_bf16(af);

    // ---- math phase: 4 back-to-back WMMAs ----
#pragma unroll
    for (int t = 0; t < 4; ++t)
      acc[t] = __builtin_amdgcn_wmma_f32_16x16x32_bf16(
          false, a, false, b[t], (short)0, acc[t], false, false);
  }

#pragma unroll
  for (int t = 0; t < 4; ++t) {
    const float bias = Bb[t * 16 + m];
#pragma unroll
    for (int g = 0; g < 8; ++g) {
      const int r = row0 + g + 8 * half;            // C/D layout: row = g (+8 for hi half)
      Out[(size_t)r * ED + t * 16 + m] = f2bf(acc[t][g] + bias);
    }
  }
}

// ---------------------------------------------------------------------------
// Kernel 2: causal flash attention over bf16 qp/kp/vp, fp32 accumulation.
// K tile staged with GLOBAL_LOAD_ASYNC_TO_LDS_B128 (ASYNCcnt path); V tile
// transposed through registers into LDS for the P*V B-operand.
// ---------------------------------------------------------------------------
__global__ __launch_bounds__(128)
void attn_kernel(const unsigned short* __restrict__ qp,
                 const unsigned short* __restrict__ kp,
                 const unsigned short* __restrict__ vp,
                 const int* __restrict__ pad,
                 float* __restrict__ out)
{
  __shared__ unsigned short ldsK[32 * 64];      // K tile [j_local][e]
  __shared__ unsigned short ldsVT[64 * 32];     // V tile transposed [e][j_local]
  __shared__ unsigned short ldsP[4][16 * 32];   // per-wave P scratch [row][col]

  const int n    = blockIdx.x;
  const int ib   = blockIdx.y;
  const int wave = threadIdx.x >> 5;
  const int lane = threadIdx.x & 31;
  const int m    = lane & 15;
  const int half = lane >> 4;
  const int i0   = ib * 64 + wave * 16;
  const size_t rb = (size_t)n * LQ;

  // Load this wave's 16x64 Q tile into A-layout registers (held for all j)
  const unsigned short* qrow = qp + (rb + i0 + m) * ED;
  v16bf aq[2];
#pragma unroll
  for (int c = 0; c < 2; ++c) {
    u16x8 lo = *(const u16x8*)(qrow + c * 32 + half * 16);
    u16x8 hi = *(const u16x8*)(qrow + c * 32 + half * 16 + 8);
    aq[c] = combine8(lo, hi);
  }

  bool padok[8];
#pragma unroll
  for (int g = 0; g < 8; ++g) padok[g] = pad[rb + i0 + g + 8 * half] != 0;

  v8f z = {0.f, 0.f, 0.f, 0.f, 0.f, 0.f, 0.f, 0.f};
  v8f o[4] = {z, z, z, z};
  float mrow[8], lrow[8];
#pragma unroll
  for (int g = 0; g < 8; ++g) { mrow[g] = -__builtin_inff(); lrow[g] = 0.f; }

  const float scale = 0.022097086912079608f;    // 1/sqrt(L=2048)
  const int nblk = (ib + 1) * 2;                // 32-wide j blocks up to block's max row

  // Per-thread staging coordinates: thread -> (row r, 16-element e chunk)
  const int sr = threadIdx.x >> 2;              // 0..31
  const int se = (threadIdx.x & 3) * 16;
  const unsigned lk = (unsigned)(unsigned long long)&ldsK[sr * 64 + se]; // LDS byte addr

  for (int jb = 0; jb < nblk; ++jb) {
    const int j0 = jb * 32;

    __syncthreads();   // previous block's LDS consumers done
    {
      // K tile: async DMA global -> LDS, 2 x 16B per thread (ASYNCcnt-tracked)
      const unsigned long long ga =
          (unsigned long long)(const void*)(kp + (rb + j0 + sr) * ED + se);
      asm volatile("global_load_async_to_lds_b128 %0, %1, off"
                   :: "v"(lk), "v"(ga) : "memory");
      asm volatile("global_load_async_to_lds_b128 %0, %1, off offset:16"
                   :: "v"(lk), "v"(ga) : "memory");

      // V tile: load to registers, transpose into LDS
      const unsigned short* vsrc = vp + (rb + j0 + sr) * ED + se;
      u16x8 v0v = *(const u16x8*)vsrc;
      u16x8 v1v = *(const u16x8*)(vsrc + 8);
#pragma unroll
      for (int e = 0; e < 8; ++e) {
        ldsVT[(se + e) * 32 + sr]     = v0v[e];
        ldsVT[(se + 8 + e) * 32 + sr] = v1v[e];
      }
      asm volatile("s_wait_asynccnt 0x0" ::: "memory");
    }
    __syncthreads();

    if (j0 <= i0 + 15) {                        // wave-uniform skip of dead blocks
      // ---- S = Qtile * Ktile^T (two 16-col tiles, K=64 in two chunks) ----
      v8f s0 = z, s1 = z;
#pragma unroll
      for (int c = 0; c < 2; ++c) {
        const unsigned short* b0p = &ldsK[m * 64 + c * 32 + half * 16];
        const unsigned short* b1p = &ldsK[(16 + m) * 64 + c * 32 + half * 16];
        v16bf b0 = combine8(*(const u16x8*)b0p, *(const u16x8*)(b0p + 8));
        v16bf b1 = combine8(*(const u16x8*)b1p, *(const u16x8*)(b1p + 8));
        s0 = __builtin_amdgcn_wmma_f32_16x16x32_bf16(false, aq[c], false, b0,
                                                     (short)0, s0, false, false);
        s1 = __builtin_amdgcn_wmma_f32_16x16x32_bf16(false, aq[c], false, b1,
                                                     (short)0, s1, false, false);
      }

      // ---- online softmax (per-row stats across 16-lane halves) ----
      unsigned short* pr = &ldsP[wave][0];
#pragma unroll
      for (int g = 0; g < 8; ++g) {
        const int r = i0 + g + 8 * half;
        float x0 = s0[g] * scale;
        float x1 = s1[g] * scale;
        if ((j0 + m > r)      || !padok[g]) x0 = -__builtin_inff();
        if ((j0 + 16 + m > r) || !padok[g]) x1 = -__builtin_inff();
        float mx = fmaxf(x0, x1);
#pragma unroll
        for (int off = 1; off < 16; off <<= 1)
          mx = fmaxf(mx, __shfl_xor(mx, off, 32));
        const float mnew  = fmaxf(mrow[g], mx);
        const float alpha = __expf(mrow[g] - mnew);
        mrow[g] = mnew;
        const float p0 = __expf(x0 - mnew);
        const float p1 = __expf(x1 - mnew);
        float ps = p0 + p1;
#pragma unroll
        for (int off = 1; off < 16; off <<= 1)
          ps += __shfl_xor(ps, off, 32);
        lrow[g] = lrow[g] * alpha + ps;
#pragma unroll
        for (int t = 0; t < 4; ++t) o[t][g] = o[t][g] * alpha;
        unsigned short* prow = pr + (g + 8 * half) * 32;   // D-layout -> [row][col]
        prow[m]      = f2bf(p0);
        prow[16 + m] = f2bf(p1);
      }

      // same-wave LDS store->load ordering (split dep counter on CDNA5)
      asm volatile("s_wait_dscnt 0x0" ::: "memory");

      // ---- re-read P in A-layout, O += P * Vtile ----
      const unsigned short* prd = pr + m * 32;
      v16bf ap = combine8(*(const u16x8*)(prd + half * 16),
                          *(const u16x8*)(prd + half * 16 + 8));
#pragma unroll
      for (int t = 0; t < 4; ++t) {
        const unsigned short* vb = &ldsVT[(t * 16 + m) * 32 + half * 16];
        v16bf bvv = combine8(*(const u16x8*)vb, *(const u16x8*)(vb + 8));
        o[t] = __builtin_amdgcn_wmma_f32_16x16x32_bf16(false, ap, false, bvv,
                                                       (short)0, o[t], false, false);
      }
    }
  }

  // ---- normalize and store fp32 output ----
#pragma unroll
  for (int t = 0; t < 4; ++t) {
#pragma unroll
    for (int g = 0; g < 8; ++g) {
      const int r = i0 + g + 8 * half;
      out[(rb + r) * ED + t * 16 + m] = o[t][g] / lrow[g];
    }
  }
}

extern "C" void kernel_launch(void* const* d_in, const int* in_sizes, int n_in,
                              void* d_out, int out_size, void* d_ws, size_t ws_size,
                              hipStream_t stream) {
  const float* q   = (const float*)d_in[0];
  const float* k   = (const float*)d_in[1];
  const float* v   = (const float*)d_in[2];
  const int*   pad = (const int*)d_in[3];
  const float* Wq  = (const float*)d_in[4];
  const float* bq  = (const float*)d_in[5];
  const float* Wk  = (const float*)d_in[6];
  const float* bk  = (const float*)d_in[7];
  const float* Wv  = (const float*)d_in[8];
  const float* bv  = (const float*)d_in[9];

  unsigned short* qp = (unsigned short*)d_ws;            // 2 MB bf16
  unsigned short* kp = qp + (size_t)NROWS * ED;          // 2 MB bf16
  unsigned short* vp = kp + (size_t)NROWS * ED;          // 2 MB bf16
  unsigned short* Wb = vp + (size_t)NROWS * ED;          // 3 x 128 KB bf16

  cvt_w_kernel<<<dim3(ED * HD / (256 * 4), 3), 256, 0, stream>>>(Wq, Wk, Wv, Wb);
  proj_kernel<<<dim3(NROWS / 64, 3), 128, 0, stream>>>(
      q, k, v, Wb, bq, bk, bv, qp, kp, vp);
  attn_kernel<<<dim3(NB, LQ / 64), 128, 0, stream>>>(
      qp, kp, vp, pad, (float*)d_out);
}
